// MultiHeadAttention_6932077216008
// MI455X (gfx1250) — matrix operand order repaired and checked by reading, not compiled, not run on hardware
//
#include <hip/hip_runtime.h>
#include <hip/hip_fp16.h>

typedef __attribute__((ext_vector_type(16))) _Float16 v16h;
typedef __attribute__((ext_vector_type(8)))  float    v8f;

#define T_    8
#define B_    2
#define H_    16
#define W_    16
#define DM    384
#define DH    512
#define NHEAD 8
#define CV    64
#define LSEQ  2048          // T*H*W
#define ROWS  4096          // T*B*H*W
#define EPS_  1e-5f
#define OUT_ELEMS (ROWS * DM)
#define NBH   (B_ * NHEAD)  // 16

// ---------------- WMMA fragment helpers (CDNA5 wave32 layouts) ----------------

// A-matrix 16x32 f16 (row-major, ld in halves)
// lane&15 = row; lanes<16: K = k0+{0..7,16..23}; lanes>=16: K = k0+{8..15,24..31}
__device__ __forceinline__ v16h load_a_f16(const _Float16* __restrict__ src, long ld,
                                           int m0, int k0, int lane) {
    const _Float16* p = src + (long)(m0 + (lane & 15)) * ld + k0 + ((lane & 16) ? 8 : 0);
    v16h a;
#pragma unroll
    for (int j = 0; j < 8; ++j) { a[j] = p[j]; a[j + 8] = p[j + 16]; }
    return a;
}

// A-matrix 16x32 from f32 source (cvt in-register)
__device__ __forceinline__ v16h load_a_f32(const float* __restrict__ src, long ld,
                                           int m0, int k0, int lane) {
    const float* p = src + (long)(m0 + (lane & 15)) * ld + k0 + ((lane & 16) ? 8 : 0);
    v16h a;
#pragma unroll
    for (int j = 0; j < 8; ++j) { a[j] = (_Float16)p[j]; a[j + 8] = (_Float16)p[j + 16]; }
    return a;
}

// B-matrix 32x16 where B[K][N] = srcT[N][K] (srcT row-major indexed by N) -> contiguous 32B reads
// lane&15 = N; lanes<16 hold K=k0+{0..7,16..23}, lanes>=16 hold K=k0+{8..15,24..31}
__device__ __forceinline__ v16h load_bT_f16(const _Float16* __restrict__ srcT, long ld,
                                            int k0, int n0, int lane) {
    const _Float16* p = srcT + (long)(n0 + (lane & 15)) * ld + k0 + ((lane & 16) ? 8 : 0);
    v16h b;
#pragma unroll
    for (int j = 0; j < 8; ++j) { b[j] = p[j]; b[j + 8] = p[j + 16]; }
    return b;
}

#define WMMA_F16(a, b, c) __builtin_amdgcn_wmma_f32_16x16x32_f16(false, (a), false, (b), (short)0, (c), false, false)

// ---------------- Kernels ----------------

// transpose-convert f32 [K][N] -> f16 [N][K]
__global__ void cvt_transpose_f16(const float* __restrict__ src, _Float16* __restrict__ dst,
                                  int K, int N) {
    int i = blockIdx.x * blockDim.x + threadIdx.x;
    if (i < K * N) {
        int k = i / N, n = i % N;
        dst[(long)n * K + k] = (_Float16)src[i];
    }
}

// C[ROWS,DH] = X_f32[ROWS,DM] x W[DM,DH] + pe (+bias); W given transposed f16 [DH][DM]
__global__ void proj_gemm(const float* __restrict__ X, const _Float16* __restrict__ WT,
                          const float* __restrict__ pe, const float* __restrict__ bias,
                          float* __restrict__ C, int useBias) {
    int lane = threadIdx.x & 31;
    int m0 = (blockIdx.x * 8 + (threadIdx.x >> 5)) << 4;   // 256 m-tiles, grid.x=32, 8 waves
    int n0 = blockIdx.y << 6;                              // 8 n-groups of 64
    v8f a0 = {}, a1 = {}, a2 = {}, a3 = {};
    for (int k0 = 0; k0 < DM; k0 += 32) {
        v16h a  = load_a_f32(X, DM, m0, k0, lane);
        v16h b0 = load_bT_f16(WT, DM, k0, n0,      lane);
        v16h b1 = load_bT_f16(WT, DM, k0, n0 + 16, lane);
        v16h b2 = load_bT_f16(WT, DM, k0, n0 + 32, lane);
        v16h b3 = load_bT_f16(WT, DM, k0, n0 + 48, lane);
        a0 = WMMA_F16(a, b0, a0);
        a1 = WMMA_F16(a, b1, a1);
        a2 = WMMA_F16(a, b2, a2);
        a3 = WMMA_F16(a, b3, a3);
    }
    v8f accs[4] = { a0, a1, a2, a3 };
    int rb = m0 + ((lane & 16) ? 8 : 0);
#pragma unroll
    for (int nt = 0; nt < 4; ++nt) {
        int col = n0 + nt * 16 + (lane & 15);
        float bv = useBias ? bias[col] : 0.0f;
#pragma unroll
        for (int r = 0; r < 8; ++r) {
            long idx = (long)(rb + r) * DH + col;
            C[idx] = accs[nt][r] + pe[idx] + bv;
        }
    }
}

// per-channel mean & 1/sqrt(var+eps) over 4096 rows
__global__ void bn_stats(const float* __restrict__ z, int C,
                         float* __restrict__ mean, float* __restrict__ invstd) {
    __shared__ float s1[256], s2[256];
    int c = blockIdx.x;
    float sum = 0.f, sq = 0.f;
    for (int r = threadIdx.x; r < ROWS; r += 256) {
        float x = z[(long)r * C + c];
        sum += x; sq += x * x;
    }
    s1[threadIdx.x] = sum; s2[threadIdx.x] = sq; __syncthreads();
    for (int off = 128; off > 0; off >>= 1) {
        if (threadIdx.x < off) { s1[threadIdx.x] += s1[threadIdx.x + off]; s2[threadIdx.x] += s2[threadIdx.x + off]; }
        __syncthreads();
    }
    if (threadIdx.x == 0) {
        float mu  = s1[0] / (float)ROWS;
        float var = s2[0] / (float)ROWS - mu * mu;
        mean[c] = mu;
        invstd[c] = rsqrtf(var + EPS_);
    }
}

// BN + IF scan over T, emit (2s-1) f16 into [b,head,L,CV]
__global__ void if_spike_heads(const float* __restrict__ z, const float* __restrict__ mean,
                               const float* __restrict__ invstd, const float* __restrict__ gamma,
                               const float* __restrict__ beta, _Float16* __restrict__ sp) {
    int tid = blockIdx.x * blockDim.x + threadIdx.x;   // B*H*W*DH = 262144
    int c = tid & 511; int rest = tid >> 9;
    int w = rest & 15; rest >>= 4;
    int h = rest & 15; int b = rest >> 4;
    int head = c >> 6, d = c & 63;
    float mu = mean[c], is = invstd[c], g = gamma[c], be = beta[c];
    int rb = b * 256 + h * 16 + w;
    long spBase = ((long)(b * NHEAD + head) * LSEQ) * CV + d;
    float v = 0.f;
    for (int t = 0; t < T_; ++t) {
        float x = (z[(long)(t * 512 + rb) * DH + c] - mu) * is * g + be;
        v += x;
        float s = (v >= 1.f) ? 1.f : 0.f;
        sp[spBase + (long)(t * 256 + h * 16 + w) * CV] = (_Float16)(2.f * s - 1.f);
        v *= (1.f - s);
    }
}

// value path to f16, TRANSPOSED per head: [b,head,CV,L]
__global__ void v_headsT(const float* __restrict__ vz, _Float16* __restrict__ spT) {
    int tid = blockIdx.x * blockDim.x + threadIdx.x;   // ROWS*DH
    int c = tid & 511; int r = tid >> 9;
    int w = r & 15, h = (r >> 4) & 15, b = (r >> 8) & 1, t = r >> 9;
    int head = c >> 6, d = c & 63;
    int l = t * 256 + h * 16 + w;
    spT[((long)(b * NHEAD + head) * CV + d) * LSEQ + l] = (_Float16)vz[(long)r * DH + c];
}

// raw scores -> d_out score region; 16x64 per wave
__global__ void score_gemm(const _Float16* __restrict__ qsp, const _Float16* __restrict__ ksp,
                           float* __restrict__ score) {
    int lane = threadIdx.x & 31;
    int m0 = (blockIdx.x * 8 + (threadIdx.x >> 5)) << 4;   // grid.x=16, 128 m-tiles
    int n0 = blockIdx.y << 6;                              // 32 n-groups
    int bh = blockIdx.z;
    const _Float16* qb = qsp + (long)bh * LSEQ * CV;
    const _Float16* kb = ksp + (long)bh * LSEQ * CV;
    v8f a0 = {}, a1 = {}, a2 = {}, a3 = {};
#pragma unroll
    for (int k0 = 0; k0 < CV; k0 += 32) {
        v16h a  = load_a_f16(qb, CV, m0, k0, lane);
        v16h b0 = load_bT_f16(kb, CV, k0, n0,      lane);
        v16h b1 = load_bT_f16(kb, CV, k0, n0 + 16, lane);
        v16h b2 = load_bT_f16(kb, CV, k0, n0 + 32, lane);
        v16h b3 = load_bT_f16(kb, CV, k0, n0 + 48, lane);
        a0 = WMMA_F16(a, b0, a0);
        a1 = WMMA_F16(a, b1, a1);
        a2 = WMMA_F16(a, b2, a2);
        a3 = WMMA_F16(a, b3, a3);
    }
    v8f accs[4] = { a0, a1, a2, a3 };
    float* outp = score + (long)bh * LSEQ * LSEQ;
    int rb = m0 + ((lane & 16) ? 8 : 0);
#pragma unroll
    for (int nt = 0; nt < 4; ++nt) {
        int col = n0 + nt * 16 + (lane & 15);
#pragma unroll
        for (int r = 0; r < 8; ++r)
            outp[(long)(rb + r) * LSEQ + col] = 0.5f * (1.f + accs[nt][r] * (1.f / 64.f));
    }
}

// in-place softmax over rows of 2048; optional f16 copy for the AV GEMM
__global__ void softmax_rows(float* __restrict__ s, _Float16* __restrict__ sh) {
    __shared__ float red[256];
    float* p = s + (long)blockIdx.x * LSEQ;
    int tid = threadIdx.x;
    float m = -3.4e38f;
    for (int i = tid; i < LSEQ; i += 256) m = fmaxf(m, p[i]);
    red[tid] = m; __syncthreads();
    for (int off = 128; off > 0; off >>= 1) {
        if (tid < off) red[tid] = fmaxf(red[tid], red[tid + off]);
        __syncthreads();
    }
    m = red[0]; __syncthreads();
    float sum = 0.f;
    for (int i = tid; i < LSEQ; i += 256) { float e = __expf(p[i] - m); p[i] = e; sum += e; }
    red[tid] = sum; __syncthreads();
    for (int off = 128; off > 0; off >>= 1) {
        if (tid < off) red[tid] += red[tid + off];
        __syncthreads();
    }
    float inv = 1.f / red[0];
    if (sh) {
        _Float16* ph = sh + (long)blockIdx.x * LSEQ;
        for (int i = tid; i < LSEQ; i += 256) {
            float vv = p[i] * inv;
            p[i] = vv;
            ph[i] = (_Float16)vv;
        }
    } else {
        for (int i = tid; i < LSEQ; i += 256) p[i] *= inv;
    }
}

__device__ __forceinline__ void attn_store(float* __restrict__ obuf, int b, int head,
                                           int m0, int n0, int lane, const v8f& acc) {
    int d  = n0 + (lane & 15);
    int lb = m0 + ((lane & 16) ? 8 : 0);
    int c  = head * CV + d;
#pragma unroll
    for (int r = 0; r < 8; ++r) {
        int l = lb + r;
        int t = l >> 8, hw = l & 255;
        obuf[(long)(t * 512 + b * 256 + hw) * DH + c] = acc[r];
    }
}

// o = softmax(score) x V, f16 score path; 16x64 per wave covers all CV
__global__ void attn_gemm_f16(const _Float16* __restrict__ scoreh, const _Float16* __restrict__ vspT,
                              float* __restrict__ obuf) {
    int lane = threadIdx.x & 31;
    int m0 = (blockIdx.x * 8 + (threadIdx.x >> 5)) << 4;   // grid.x=16
    int bh = blockIdx.z;
    int b = bh >> 3, head = bh & 7;
    const _Float16* sc = scoreh + (long)bh * LSEQ * LSEQ;
    const _Float16* vb = vspT   + (long)bh * CV * LSEQ;
    v8f a0 = {}, a1 = {}, a2 = {}, a3 = {};
    for (int k0 = 0; k0 < LSEQ; k0 += 32) {
        __builtin_prefetch(sc + (long)(m0 + (lane & 15)) * LSEQ + k0 + 128, 0, 0);
        v16h a  = load_a_f16(sc, LSEQ, m0, k0, lane);
        v16h b0 = load_bT_f16(vb, LSEQ, k0, 0,  lane);
        v16h b1 = load_bT_f16(vb, LSEQ, k0, 16, lane);
        v16h b2 = load_bT_f16(vb, LSEQ, k0, 32, lane);
        v16h b3 = load_bT_f16(vb, LSEQ, k0, 48, lane);
        a0 = WMMA_F16(a, b0, a0);
        a1 = WMMA_F16(a, b1, a1);
        a2 = WMMA_F16(a, b2, a2);
        a3 = WMMA_F16(a, b3, a3);
    }
    attn_store(obuf, b, head, m0, 0,  lane, a0);
    attn_store(obuf, b, head, m0, 16, lane, a1);
    attn_store(obuf, b, head, m0, 32, lane, a2);
    attn_store(obuf, b, head, m0, 48, lane, a3);
}

// fallback: f32 scores converted in-register
__global__ void attn_gemm_f32(const float* __restrict__ score, const _Float16* __restrict__ vspT,
                              float* __restrict__ obuf) {
    int lane = threadIdx.x & 31;
    int m0 = (blockIdx.x * 8 + (threadIdx.x >> 5)) << 4;
    int bh = blockIdx.z;
    int b = bh >> 3, head = bh & 7;
    const float*    sc = score + (long)bh * LSEQ * LSEQ;
    const _Float16* vb = vspT  + (long)bh * CV * LSEQ;
    v8f a0 = {}, a1 = {}, a2 = {}, a3 = {};
    for (int k0 = 0; k0 < LSEQ; k0 += 32) {
        __builtin_prefetch(sc + (long)(m0 + (lane & 15)) * LSEQ + k0 + 64, 0, 0);
        v16h a  = load_a_f32(sc, LSEQ, m0, k0, lane);
        v16h b0 = load_bT_f16(vb, LSEQ, k0, 0,  lane);
        v16h b1 = load_bT_f16(vb, LSEQ, k0, 16, lane);
        v16h b2 = load_bT_f16(vb, LSEQ, k0, 32, lane);
        v16h b3 = load_bT_f16(vb, LSEQ, k0, 48, lane);
        a0 = WMMA_F16(a, b0, a0);
        a1 = WMMA_F16(a, b1, a1);
        a2 = WMMA_F16(a, b2, a2);
        a3 = WMMA_F16(a, b3, a3);
    }
    attn_store(obuf, b, head, m0, 0,  lane, a0);
    attn_store(obuf, b, head, m0, 16, lane, a1);
    attn_store(obuf, b, head, m0, 32, lane, a2);
    attn_store(obuf, b, head, m0, 48, lane, a3);
}

// BN + IF, 0/1 spikes f16 row-major [ROWS,DH]
__global__ void if_spike_plain(const float* __restrict__ z, const float* __restrict__ mean,
                               const float* __restrict__ invstd, const float* __restrict__ gamma,
                               const float* __restrict__ beta, _Float16* __restrict__ sp) {
    int tid = blockIdx.x * blockDim.x + threadIdx.x;
    int c = tid & 511; int rest = tid >> 9;
    int w = rest & 15; rest >>= 4;
    int h = rest & 15; int b = rest >> 4;
    int rb = b * 256 + h * 16 + w;
    float mu = mean[c], is = invstd[c], g = gamma[c], be = beta[c];
    float v = 0.f;
    for (int t = 0; t < T_; ++t) {
        long idx = (long)(t * 512 + rb) * DH + c;
        float x = (z[idx] - mu) * is * g + be;
        v += x;
        float s = (v >= 1.f) ? 1.f : 0.f;
        sp[idx] = (_Float16)s;
        v *= (1.f - s);
    }
}

// zout[ROWS,DM] = osp[ROWS,DH] x Wout[DH,DM] + bout ; Wout transposed f16 [DM][DH]
__global__ void out_gemm(const _Float16* __restrict__ osp, const _Float16* __restrict__ WoT,
                         const float* __restrict__ bout, float* __restrict__ zout) {
    int lane = threadIdx.x & 31;
    int m0 = (blockIdx.x * 8 + (threadIdx.x >> 5)) << 4;   // grid.x=32
    int n0 = blockIdx.y << 6;                              // 6 n-groups
    v8f a0 = {}, a1 = {}, a2 = {}, a3 = {};
    for (int k0 = 0; k0 < DH; k0 += 32) {
        v16h a  = load_a_f16(osp, DH, m0, k0, lane);
        v16h b0 = load_bT_f16(WoT, DH, k0, n0,      lane);
        v16h b1 = load_bT_f16(WoT, DH, k0, n0 + 16, lane);
        v16h b2 = load_bT_f16(WoT, DH, k0, n0 + 32, lane);
        v16h b3 = load_bT_f16(WoT, DH, k0, n0 + 48, lane);
        a0 = WMMA_F16(a, b0, a0);
        a1 = WMMA_F16(a, b1, a1);
        a2 = WMMA_F16(a, b2, a2);
        a3 = WMMA_F16(a, b3, a3);
    }
    v8f accs[4] = { a0, a1, a2, a3 };
    int rb = m0 + ((lane & 16) ? 8 : 0);
#pragma unroll
    for (int nt = 0; nt < 4; ++nt) {
        int col = n0 + nt * 16 + (lane & 15);
        float bv = bout[col];
#pragma unroll
        for (int r = 0; r < 8; ++r)
            zout[(long)(rb + r) * DM + col] = accs[nt][r] + bv;
    }
}

// final BN + IF -> f32 spikes to d_out [T,B,H,W,DM]
__global__ void if_final(const float* __restrict__ z, const float* __restrict__ mean,
                         const float* __restrict__ invstd, const float* __restrict__ gamma,
                         const float* __restrict__ beta, float* __restrict__ out) {
    int tid = blockIdx.x * blockDim.x + threadIdx.x;   // B*H*W*DM
    int c = tid % DM; int rest = tid / DM;
    int w = rest & 15; rest >>= 4;
    int h = rest & 15; int b = rest >> 4;
    int rb = b * 256 + h * 16 + w;
    float mu = mean[c], is = invstd[c], g = gamma[c], be = beta[c];
    float v = 0.f;
    for (int t = 0; t < T_; ++t) {
        long idx = (long)(t * 512 + rb) * DM + c;
        float x = (z[idx] - mu) * is * g + be;
        v += x;
        float s = (v >= 1.f) ? 1.f : 0.f;
        out[idx] = s;
        v *= (1.f - s);
    }
}

// ---------------- Host launcher ----------------

extern "C" void kernel_launch(void* const* d_in, const int* in_sizes, int n_in,
                              void* d_out, int out_size, void* d_ws, size_t ws_size,
                              hipStream_t stream) {
    const float* q    = (const float*)d_in[0];
    const float* k    = (const float*)d_in[1];
    const float* v    = (const float*)d_in[2];
    const float* pe   = (const float*)d_in[3];
    const float* Wq   = (const float*)d_in[4];
    const float* Wk   = (const float*)d_in[5];
    const float* Wv   = (const float*)d_in[6];
    const float* bv   = (const float*)d_in[7];
    const float* Wout = (const float*)d_in[8];
    const float* bout = (const float*)d_in[9];
    const float* g_q  = (const float*)d_in[10];
    const float* b_q  = (const float*)d_in[11];
    const float* g_k  = (const float*)d_in[12];
    const float* b_k  = (const float*)d_in[13];
    const float* g_v  = (const float*)d_in[14];
    const float* b_v  = (const float*)d_in[15];
    const float* g_o  = (const float*)d_in[16];
    const float* b_o  = (const float*)d_in[17];

    float* out_final = (float*)d_out;
    float* score     = (float*)d_out + OUT_ELEMS;

    char* p = (char*)d_ws;
    auto carve = [&](size_t bytes) { void* r = (void*)p; p += (bytes + 255) & ~(size_t)255; return r; };
    _Float16* WqT = (_Float16*)carve((size_t)DM * DH * 2);   // [DH][DM]
    _Float16* WkT = (_Float16*)carve((size_t)DM * DH * 2);
    _Float16* WvT = (_Float16*)carve((size_t)DM * DH * 2);
    _Float16* WoT = (_Float16*)carve((size_t)DH * DM * 2);   // [DM][DH]
    float* qz   = (float*)carve((size_t)ROWS * DH * 4);
    float* kz   = (float*)carve((size_t)ROWS * DH * 4);
    float* vz   = (float*)carve((size_t)ROWS * DH * 4);
    float* obuf = (float*)carve((size_t)ROWS * DH * 4);
    float* zout = (float*)carve((size_t)ROWS * DM * 4);
    _Float16* qsp  = (_Float16*)carve((size_t)NBH * LSEQ * CV * 2);
    _Float16* ksp  = (_Float16*)carve((size_t)NBH * LSEQ * CV * 2);
    _Float16* vspT = (_Float16*)carve((size_t)NBH * CV * LSEQ * 2);
    _Float16* osp  = (_Float16*)carve((size_t)ROWS * DH * 2);
    float* meanQ = (float*)carve(DH * 4);  float* invQ = (float*)carve(DH * 4);
    float* meanK = (float*)carve(DH * 4);  float* invK = (float*)carve(DH * 4);
    float* meanO = (float*)carve(DH * 4);  float* invO = (float*)carve(DH * 4);
    float* meanF = (float*)carve(DM * 4);  float* invF = (float*)carve(DM * 4);

    // f16 score copy only if the workspace can hold it (saves 128 MB of HBM reads on AV GEMM)
    size_t used = (size_t)(p - (char*)d_ws);
    size_t scoreh_bytes = (size_t)NBH * LSEQ * LSEQ * 2;
    _Float16* scoreh = (used + scoreh_bytes <= ws_size) ? (_Float16*)p : nullptr;

    // 1) weights -> transposed f16
    cvt_transpose_f16<<<(DM * DH + 255) / 256, 256, 0, stream>>>(Wq,   WqT, DM, DH);
    cvt_transpose_f16<<<(DM * DH + 255) / 256, 256, 0, stream>>>(Wk,   WkT, DM, DH);
    cvt_transpose_f16<<<(DM * DH + 255) / 256, 256, 0, stream>>>(Wv,   WvT, DM, DH);
    cvt_transpose_f16<<<(DH * DM + 255) / 256, 256, 0, stream>>>(Wout, WoT, DH, DM);

    // 2) projections (+PE, +bias for V)
    {
        dim3 g(32, 8);
        proj_gemm<<<g, 256, 0, stream>>>(q, WqT, pe, bv, qz, 0);
        proj_gemm<<<g, 256, 0, stream>>>(k, WkT, pe, bv, kz, 0);
        proj_gemm<<<g, 256, 0, stream>>>(v, WvT, pe, bv, vz, 1);
    }

    // 3) BN stats + IF spikes; V -> transposed f16 heads
    bn_stats<<<DH, 256, 0, stream>>>(qz, DH, meanQ, invQ);
    bn_stats<<<DH, 256, 0, stream>>>(kz, DH, meanK, invK);
    if_spike_heads<<<1024, 256, 0, stream>>>(qz, meanQ, invQ, g_q, b_q, qsp);
    if_spike_heads<<<1024, 256, 0, stream>>>(kz, meanK, invK, g_k, b_k, ksp);
    v_headsT<<<(ROWS * DH) / 256, 256, 0, stream>>>(vz, vspT);

    // 4) scores -> d_out, softmax in place (+optional f16 copy)
    {
        dim3 g(16, 32, NBH);
        score_gemm<<<g, 256, 0, stream>>>(qsp, ksp, score);
    }
    softmax_rows<<<NBH * LSEQ, 256, 0, stream>>>(score, scoreh);

    // 5) A x V
    {
        dim3 g(16, 1, NBH);
        if (scoreh) attn_gemm_f16<<<g, 256, 0, stream>>>(scoreh, vspT, obuf);
        else        attn_gemm_f32<<<g, 256, 0, stream>>>(score, vspT, obuf);
    }
    bn_stats<<<DH, 256, 0, stream>>>(obuf, DH, meanO, invO);
    if_spike_plain<<<1024, 256, 0, stream>>>(obuf, meanO, invO, g_v, b_v, osp);

    // 6) output projection, BN + IF -> final output
    {
        dim3 g(32, 6);
        out_gemm<<<g, 256, 0, stream>>>(osp, WoT, bout, zout);
    }
    bn_stats<<<DM, 256, 0, stream>>>(zout, DM, meanF, invF);
    if_final<<<768, 256, 0, stream>>>(zout, meanF, invF, g_o, b_o, out_final);
}
